// MyGo_Loss_83880711291624
// MI455X (gfx1250) — 8-bit matrix operand order repaired and checked by reading, compiled, not run on hardware
//
#include <hip/hip_runtime.h>
#include <stdint.h>

#define L_ 6
#define B_ 8
#define N_ 300
#define C_ 20
#define HW_ 784
#define NP_ 320
#define KP_ 832
#define GT_ 448

typedef __attribute__((ext_vector_type(8))) int v8i;

static const size_t RM_OFF  = 0;
static const size_t T_OFF   = RM_OFF  + (size_t)B_*N_*HW_*4;
static const size_t S_OFF   = T_OFF   + (size_t)B_*NP_*KP_;
static const size_t ADJ_OFF = S_OFF   + (size_t)B_*NP_*4;
static const size_t FIN_OFF = ADJ_OFF + (size_t)B_*NP_*NP_;
static const size_t CNT_OFF = FIN_OFF + (size_t)B_*C_*10*4;
static const size_t MM_OFF  = CNT_OFF + (size_t)B_*C_*4;
static const size_t ACC_OFF = MM_OFF  + (size_t)B_*C_*HW_*4;

__device__ __forceinline__ int clampi(int v, int lo, int hi) {
  return v < lo ? lo : (v > hi ? hi : v);
}

__global__ void prep_kernel(const float* __restrict__ score_map, char* ws) {
  float*   rm = (float*)(ws + RM_OFF);
  uint8_t* T  = (uint8_t*)(ws + T_OFF);
  float*   s  = (float*)(ws + S_OFF);
  float*   accum = (float*)(ws + ACC_OFF);

  const int b   = blockIdx.x / NP_;
  const int n   = blockIdx.x % NP_;
  const int tid = threadIdx.x;
  if (blockIdx.x == 0 && tid == 0) accum[0] = 0.f;

  __shared__ float red[256];
  uint8_t* Trow = T + ((size_t)b*NP_ + n) * KP_;

  if (n >= N_) {
    for (int i = tid; i < KP_; i += 256) Trow[i] = 0;
    if (tid == 0) s[b*NP_ + n] = 0.f;
    return;
  }

  float v[4];
  #pragma unroll
  for (int i = 0; i < 4; ++i) {
    int hw = tid + i * 256;
    float acc = 0.f;
    if (hw < HW_) {
      #pragma unroll
      for (int l = 0; l < L_; ++l)
        acc += score_map[(((size_t)l*B_ + b)*N_ + n)*HW_ + hw];
    }
    v[i] = fmaxf(acc * (1.0f / L_), 0.f);
  }
  float mx = fmaxf(fmaxf(v[0], v[1]), fmaxf(v[2], v[3]));
  red[tid] = mx; __syncthreads();
  for (int st = 128; st > 0; st >>= 1) {
    if (tid < st) red[tid] = fmaxf(red[tid], red[tid + st]);
    __syncthreads();
  }
  mx = red[0] + 1e-6f;
  __syncthreads();

  float cnt = 0.f;
  float* rmrow = rm + ((size_t)b*N_ + n)*HW_;
  #pragma unroll
  for (int i = 0; i < 4; ++i) {
    int hw = tid + i * 256;
    if (hw < HW_) {
      float nv = v[i] / mx;
      rmrow[hw] = nv;
      uint8_t t = nv > 0.7f ? 1 : 0;
      Trow[hw] = t;
      cnt += (float)t;
    }
  }
  for (int i = HW_ + tid; i < KP_; i += 256) Trow[i] = 0;

  red[tid] = cnt; __syncthreads();
  for (int st = 128; st > 0; st >>= 1) {
    if (tid < st) red[tid] += red[tid + st];
    __syncthreads();
  }
  if (tid == 0) s[b*NP_ + n] = red[0];
}

__global__ void wmma_adj_kernel(char* ws) {
  const uint8_t* T = (const uint8_t*)(ws + T_OFF);
  const float*   s = (const float*)(ws + S_OFF);
  uint8_t*     adj = (uint8_t*)(ws + ADJ_OFF);

  const int lane = threadIdx.x & 31;
  const int wave = threadIdx.x >> 5;
  const int tile = blockIdx.x * 8 + wave;
  const int b  = tile / 400;
  const int t  = tile % 400;
  const int m0 = (t / 20) * 16;
  const int n0 = (t % 20) * 16;

  const int  l15 = lane & 15;
  const bool hiL = (lane & 16) != 0;

  const uint8_t* Arow = T + ((size_t)b*NP_ + m0 + l15) * KP_ + (hiL ? 32 : 0);
  const uint8_t* Brow = T + ((size_t)b*NP_ + n0 + l15) * KP_ + (hiL ? 32 : 0);

  v8i acc = {0, 0, 0, 0, 0, 0, 0, 0};
  for (int k0 = 0; k0 < KP_; k0 += 64) {
    v8i a, bb;
    #pragma unroll
    for (int ch = 0; ch < 4; ++ch) {
      uint2 d = *(const uint2*)(Arow + k0 + ch * 8);
      a[2*ch]     = (int)d.x;
      a[2*ch + 1] = (int)d.y;
    }
    uint4 lo = *(const uint4*)(Brow + k0);
    uint4 hh = *(const uint4*)(Brow + k0 + 16);
    bb[0] = (int)lo.x; bb[1] = (int)lo.y; bb[2] = (int)lo.z; bb[3] = (int)lo.w;
    bb[4] = (int)hh.x; bb[5] = (int)hh.y; bb[6] = (int)hh.z; bb[7] = (int)hh.w;
    acc = __builtin_amdgcn_wmma_i32_16x16x64_iu8(false, a, false, bb, acc,
                                                 false, false);
  }

  const int   n  = n0 + l15;
  const float sn = s[b*NP_ + n];
  #pragma unroll
  for (int r = 0; r < 8; ++r) {
    int   m     = m0 + r + (hiL ? 8 : 0);
    float inter = (float)acc[r];
    float sm    = s[b*NP_ + m];
    float uni   = sm + sn - inter;
    float tot   = sm + sn;
    float iou   = (uni <= 0.f) ? 1.f : inter / uni;
    float dice  = (tot <= 0.f) ? 1.f : 2.f * inter / tot;
    float Sv    = 0.5f * (iou + dice);
    adj[((size_t)b*NP_ + m)*NP_ + n] = (Sv > 0.5f) ? 1 : 0;
  }
}

__global__ void cluster_kernel(const float* __restrict__ score, char* ws) {
  const uint8_t* adj = (const uint8_t*)(ws + ADJ_OFF);
  uint32_t* finout = (uint32_t*)(ws + FIN_OFF);
  float*    cntout = (float*)(ws + CNT_OFF);

  const int b   = blockIdx.x / C_;
  const int c   = blockIdx.x % C_;
  const int tid = threadIdx.x;

  __shared__ uint32_t ab[NP_ * 10];
  __shared__ uint32_t cw[10], fw[10], nw[10];
  __shared__ int seedsh, hassh, chg;

  for (int idx = tid; idx < NP_ * 10; idx += 320) {
    int i = idx / 10, w = idx % 10;
    const uint8_t* p = adj + ((size_t)b*NP_ + i)*NP_ + w*32;
    uint32_t word = 0;
    #pragma unroll
    for (int q = 0; q < 2; ++q) {
      uint4 d = *(const uint4*)(p + q * 16);
      uint32_t vs[4] = {d.x, d.y, d.z, d.w};
      #pragma unroll
      for (int j = 0; j < 4; ++j)
        #pragma unroll
        for (int bb2 = 0; bb2 < 4; ++bb2)
          if ((vs[j] >> (8 * bb2)) & 0xffu) word |= 1u << (q*16 + j*4 + bb2);
    }
    ab[idx] = word;
  }
  if (tid < 10) cw[tid] = 0;
  __syncthreads();
  if (tid < N_) {
    float sc = score[((size_t)b*N_ + tid)*C_ + c];
    if (sc > 0.01f) atomicOr(&cw[tid >> 5], 1u << (tid & 31));
  }
  __syncthreads();
  if (tid == 0) {
    float bc = -1e30f; int seed = -1;
    float ba = -1e30f; int am = 0;
    for (int n2 = 0; n2 < N_; ++n2) {
      float sc = score[((size_t)b*N_ + n2)*C_ + c];
      if (sc > ba) { ba = sc; am = n2; }
      if (sc > 0.01f && sc > bc) { bc = sc; seed = n2; }
    }
    seedsh = (seed >= 0) ? seed : am;
    hassh  = (seed >= 0) ? 1 : 0;
  }
  __syncthreads();
  const int seed = seedsh;
  const int has  = hassh;
  if (tid < 10) {
    if (has) fw[tid] = cw[tid] & ab[seed*10 + tid];
    else     fw[tid] = ((seed >> 5) == tid) ? (1u << (seed & 31)) : 0u;
  }
  __syncthreads();

  if (has) {
    for (int it = 0; it < NP_; ++it) {
      if (tid == 0) chg = 0;
      __syncthreads();
      if (tid < 10) {
        uint32_t acc = 0;
        for (int w2 = 0; w2 < 10; ++w2) {
          uint32_t m = fw[w2];
          while (m) {
            int i = w2*32 + __builtin_ctz(m);
            acc |= ab[i*10 + tid];
            m &= m - 1;
          }
        }
        nw[tid] = cw[tid] & acc;
      }
      __syncthreads();
      if (tid < 10 && nw[tid] != fw[tid]) { fw[tid] = nw[tid]; atomicOr(&chg, 1); }
      __syncthreads();
      int done = (chg == 0);
      __syncthreads();
      if (done) break;
    }
  }
  if (tid < 10) finout[(size_t)blockIdx.x*10 + tid] = fw[tid];
  if (tid == 0) {
    int cnt = 0;
    for (int w = 0; w < 10; ++w) cnt += __popc(fw[w]);
    cntout[blockIdx.x] = (float)(cnt > 0 ? cnt : 1);
  }
}

__global__ void meanmap_kernel(char* ws) {
  const float*    rm  = (const float*)(ws + RM_OFF);
  const uint32_t* fin = (const uint32_t*)(ws + FIN_OFF);
  const float*    cnt = (const float*)(ws + CNT_OFF);
  float*          mm  = (float*)(ws + MM_OFF);

  const int bc  = blockIdx.x;
  const int b   = bc / C_;
  const int tid = threadIdx.x;

  uint32_t fw[10];
  #pragma unroll
  for (int w = 0; w < 10; ++w) fw[w] = fin[(size_t)bc*10 + w];
  const float inv = 1.0f / cnt[bc];

  for (int pix = tid; pix < HW_; pix += 256) {
    float acc = 0.f;
    #pragma unroll
    for (int w = 0; w < 10; ++w) {
      uint32_t m = fw[w];
      while (m) {
        int n = w*32 + __builtin_ctz(m);
        acc += rm[((size_t)b*N_ + n)*HW_ + pix];
        m &= m - 1;
      }
    }
    mm[(size_t)bc*HW_ + pix] = acc * inv;
  }
}

__global__ void bce_kernel(const int* __restrict__ label,
                           const int* __restrict__ gt, char* ws) {
  const float* mmg   = (const float*)(ws + MM_OFF);
  float*       accum = (float*)(ws + ACC_OFF);

  const int bc    = blockIdx.x / 98;
  const int chunk = blockIdx.x % 98;
  const int b = bc / C_;
  const int c = bc % C_;
  if (label[b*C_ + c] == 0) return;

  __shared__ float mm[HW_];
  __shared__ float red[256];
  const int tid = threadIdx.x;
  for (int i = tid; i < HW_; i += 256) mm[i] = mmg[(size_t)bc*HW_ + i];
  __syncthreads();

  float lsum = 0.f;
  #pragma unroll
  for (int j = 0; j < 8; ++j) {
    int p  = chunk*2048 + j*256 + tid;
    int oy = p / GT_;
    int ox = p % GT_;
    float fy = (oy + 0.5f) * 0.0625f - 0.5f;
    float fx = (ox + 0.5f) * 0.0625f - 0.5f;
    float yf = floorf(fy), xf = floorf(fx);
    float ty = fy - yf,    tx = fx - xf;
    int y0 = clampi((int)yf, 0, 27),     y1 = clampi((int)yf + 1, 0, 27);
    int x0 = clampi((int)xf, 0, 27),     x1 = clampi((int)xf + 1, 0, 27);
    float p00 = mm[y0*28 + x0], p01 = mm[y0*28 + x1];
    float p10 = mm[y1*28 + x0], p11 = mm[y1*28 + x1];
    float pred = (1.f - ty) * ((1.f - tx) * p00 + tx * p01)
               +        ty  * ((1.f - tx) * p10 + tx * p11);
    float yv = (gt[((size_t)b*GT_ + oy)*GT_ + ox] == c + 1) ? 1.f : 0.f;
    float lp = (pred > 0.f)       ? __logf(fmaxf(pred,       1e-38f)) : -100.f;
    float lq = (1.f - pred > 0.f) ? __logf(fmaxf(1.f - pred, 1e-38f)) : -100.f;
    lsum -= yv * lp + (1.f - yv) * lq;
  }
  red[tid] = lsum; __syncthreads();
  for (int st = 128; st > 0; st >>= 1) {
    if (tid < st) red[tid] += red[tid + st];
    __syncthreads();
  }
  if (tid == 0) atomicAdd(accum, red[0] * (1.0f / 200704.0f));
}

__global__ void final_kernel(const int* __restrict__ label, const char* ws,
                             float* __restrict__ out) {
  const float* accum = (const float*)(ws + ACC_OFF);
  __shared__ float red[256];
  int tid = threadIdx.x;
  float a = (tid < B_*C_ && label[tid] != 0) ? 1.f : 0.f;
  red[tid] = a; __syncthreads();
  for (int st = 128; st > 0; st >>= 1) {
    if (tid < st) red[tid] += red[tid + st];
    __syncthreads();
  }
  if (tid == 0) out[0] = accum[0] / red[0];
}

extern "C" void kernel_launch(void* const* d_in, const int* in_sizes, int n_in,
                              void* d_out, int out_size, void* d_ws, size_t ws_size,
                              hipStream_t stream) {
  const float* score_map = (const float*)d_in[0];
  const float* score     = (const float*)d_in[1];
  const int*   label     = (const int*)d_in[2];
  const int*   gt        = (const int*)d_in[3];
  char*  ws  = (char*)d_ws;
  float* out = (float*)d_out;

  prep_kernel    <<<dim3(B_ * NP_),   dim3(256), 0, stream>>>(score_map, ws);
  wmma_adj_kernel<<<dim3(400),        dim3(256), 0, stream>>>(ws);
  cluster_kernel <<<dim3(B_ * C_),    dim3(320), 0, stream>>>(score, ws);
  meanmap_kernel <<<dim3(B_ * C_),    dim3(256), 0, stream>>>(ws);
  bce_kernel     <<<dim3(B_ * C_ * 98), dim3(256), 0, stream>>>(label, gt, ws);
  final_kernel   <<<dim3(1),          dim3(256), 0, stream>>>(label, ws, out);
}
